// Experts_39264591020716
// MI455X (gfx1250) — matrix operand order repaired and checked by reading, not compiled, not run on hardware
//
#include <hip/hip_runtime.h>

// ---------------------------------------------------------------------------
// MoE expert FFN: act = silu(X W_gate) * (X W_up);  out = act W_down
// E=8, C=1024, H=2048, I=4096, f32 in/out.
// bf16 WMMA (v_wmma_f32_16x16x32_bf16) with f32 accumulation.
//   Kernel 1: fused gate/up GEMM + SiLU-GLU -> bf16 act in d_ws (64 MB).
//   Kernel 2: down GEMM (async LDS copies for bf16 A tile when available).
// Double-buffered LDS, one barrier per K-tile, packed bf16 conversion.
// ---------------------------------------------------------------------------

constexpr int E_ = 8;
constexpr int C_ = 1024;
constexpr int H_ = 2048;
constexpr int I_ = 4096;

typedef __attribute__((ext_vector_type(16))) __bf16 v16bf;
typedef __attribute__((ext_vector_type(8)))  __bf16 v8bf;
typedef __attribute__((ext_vector_type(8)))  float  v8f;
typedef __attribute__((ext_vector_type(4)))  float  v4f;
typedef __attribute__((ext_vector_type(4)))  int    v4i;

#if __has_builtin(__builtin_amdgcn_global_load_async_to_lds_b128) && \
    __has_builtin(__builtin_amdgcn_s_wait_asynccnt)
#define USE_ASYNC_LDS 1
#else
#define USE_ASYNC_LDS 0
#endif

#if USE_ASYNC_LDS
typedef __attribute__((address_space(1))) v4i g_v4i;   // global (AS1) int4
typedef __attribute__((address_space(3))) v4i l_v4i;   // LDS (AS3) int4
__device__ __forceinline__ g_v4i* to_gbl(const void* p) {
  return (g_v4i*)(unsigned long long)p;
}
__device__ __forceinline__ l_v4i* to_lds(void* p) {
  return (l_v4i*)(unsigned)(unsigned long long)p;  // low 32 bits = LDS offset
}
#endif

// pack two f32 into two bf16 (lo -> [15:0], hi -> [31:16]), RNE
__device__ __forceinline__ unsigned pk2bf(float lo, float hi) {
#if __has_builtin(__builtin_amdgcn_cvt_pk_bf16_f32)
  typedef __attribute__((ext_vector_type(2))) __bf16 v2bf;
  v2bf r = __builtin_amdgcn_cvt_pk_bf16_f32(lo, hi);
  return __builtin_bit_cast(unsigned, r);
#else
  unsigned a = __builtin_bit_cast(unsigned, lo);
  unsigned b = __builtin_bit_cast(unsigned, hi);
  a += 0x7fffu + ((a >> 16) & 1u);
  b += 0x7fffu + ((b >> 16) & 1u);
  return (a >> 16) | (b & 0xffff0000u);
#endif
}

__device__ __forceinline__ v16bf cat8(v8bf lo, v8bf hi) {
  return __builtin_shufflevector(lo, hi, 0, 1, 2, 3, 4, 5, 6, 7,
                                 8, 9, 10, 11, 12, 13, 14, 15);
}

// ---------------------------------------------------------------------------
// Kernel 1: gate/up GEMM + SiLU-GLU.
// Grid (I/64, C/128, E); block tile 128 x (64 gate + 64 up); 8 waves 4x2.
// ---------------------------------------------------------------------------
__global__ __launch_bounds__(256)
void gemm_gate_up_silu(const float* __restrict__ hs,
                       const float* __restrict__ wgu,
                       __bf16* __restrict__ act) {
  const int nblk = blockIdx.x;
  const int cblk = blockIdx.y;
  const int e    = blockIdx.z;

  constexpr int SA = 72;
  constexpr int SB = 72;
  __shared__ __align__(16) __bf16 AsB[2][128 * SA];
  __shared__ __align__(16) __bf16 BgB[2][64 * SB];
  __shared__ __align__(16) __bf16 BuB[2][64 * SB];

  const int tid  = threadIdx.x;
  const int lane = tid & 31;
  const int wave = tid >> 5;
  const int wm   = wave >> 1;    // 0..3
  const int wn   = wave & 1;     // 0..1

  const float* hsE = hs  + (size_t)e * C_ * H_ + (size_t)(cblk * 128) * H_;
  const float* wgE = wgu + (size_t)e * H_ * (2 * I_) + nblk * 64;
  const float* wuE = wgE + I_;

  v8f accg[2][2] = {};
  v8f accu[2][2] = {};

  // ---- staging: f32 -> bf16, A row-major [m][k], B transposed [n][k] ------
  auto stage = [&](int k0, __bf16* As, __bf16* Bg, __bf16* Bu) {
    #pragma unroll
    for (int it = 0; it < 8; ++it) {
      int t   = tid + it * 256;
      int row = t >> 4;
      int c4  = (t & 15) << 2;
      v4f v = *(const v4f*)(hsE + (size_t)row * H_ + k0 + c4);
      uint2 pk;
      pk.x = pk2bf(v[0], v[1]);
      pk.y = pk2bf(v[2], v[3]);
      *(uint2*)(&As[row * SA + c4]) = pk;
    }
    int kp = tid >> 4;          // k-pair index 0..15
    int n4 = (tid & 15) << 2;   // 0..60
    #pragma unroll
    for (int kk = 0; kk < 32; kk += 16) {
      int k = (kp + kk) << 1;   // even k in 0..62
      v4f g0 = *(const v4f*)(wgE + (size_t)(k0 + k) * (2 * I_) + n4);
      v4f g1 = *(const v4f*)(wgE + (size_t)(k0 + k + 1) * (2 * I_) + n4);
      v4f u0 = *(const v4f*)(wuE + (size_t)(k0 + k) * (2 * I_) + n4);
      v4f u1 = *(const v4f*)(wuE + (size_t)(k0 + k + 1) * (2 * I_) + n4);
      #pragma unroll
      for (int i = 0; i < 4; ++i) {
        *(unsigned*)(&Bg[(n4 + i) * SB + k]) = pk2bf(g0[i], g1[i]);
        *(unsigned*)(&Bu[(n4 + i) * SB + k]) = pk2bf(u0[i], u1[i]);
      }
    }
  };

  stage(0, AsB[0], BgB[0], BuB[0]);

  for (int k0 = 0; k0 < H_; k0 += 64) {
    const int p = (k0 >> 6) & 1;
    __syncthreads();
    if (k0 + 64 < H_) {
      stage(k0 + 64, AsB[p ^ 1], BgB[p ^ 1], BuB[p ^ 1]);
      if (k0 + 128 < H_)   // prefetch dominant weight stream two tiles ahead
        __builtin_prefetch(wgE + (size_t)(k0 + 128 + (tid >> 3)) * (2 * I_) +
                               ((tid & 7) << 3), 0, 1);
    }
    const __bf16* As = AsB[p];
    const __bf16* Bg = BgB[p];
    const __bf16* Bu = BuB[p];
    #pragma unroll
    for (int kk = 0; kk < 2; ++kk) {
      v16bf af[2];
      #pragma unroll
      for (int mt = 0; mt < 2; ++mt) {
        int row = wm * 32 + mt * 16 + (lane & 15);
        int kb  = kk * 32 + ((lane >> 4) << 4);
        af[mt] = cat8(*(const v8bf*)(&As[row * SA + kb]),
                      *(const v8bf*)(&As[row * SA + kb + 8]));
      }
      #pragma unroll
      for (int nt = 0; nt < 2; ++nt) {
        int col = wn * 32 + nt * 16 + (lane & 15);
        int kb2 = kk * 32 + ((lane >> 4) << 4);
        v16bf bg = cat8(*(const v8bf*)(&Bg[col * SB + kb2]),
                        *(const v8bf*)(&Bg[col * SB + kb2 + 8]));
        v16bf bu = cat8(*(const v8bf*)(&Bu[col * SB + kb2]),
                        *(const v8bf*)(&Bu[col * SB + kb2 + 8]));
        #pragma unroll
        for (int mt = 0; mt < 2; ++mt) {
          accg[mt][nt] = __builtin_amdgcn_wmma_f32_16x16x32_bf16(
              false, af[mt], false, bg, (short)0, accg[mt][nt], false, false);
          accu[mt][nt] = __builtin_amdgcn_wmma_f32_16x16x32_bf16(
              false, af[mt], false, bu, (short)0, accu[mt][nt], false, false);
        }
      }
    }
  }

  // ---- epilogue: silu(gate)*up -> bf16 activation -------------------------
  const int mbase = cblk * 128 + wm * 32;
  const int nbase = nblk * 64 + wn * 32;
  __bf16* actE = act + (size_t)e * C_ * I_;
  #pragma unroll
  for (int mt = 0; mt < 2; ++mt) {
    #pragma unroll
    for (int nt = 0; nt < 2; ++nt) {
      #pragma unroll
      for (int v = 0; v < 8; ++v) {
        int m = mbase + mt * 16 + v + ((lane >> 4) << 3);
        int n = nbase + nt * 16 + (lane & 15);
        float g = accg[mt][nt][v];
        float u = accu[mt][nt][v];
        float s = g / (1.0f + __expf(-g));   // SiLU
        unsigned pk = pk2bf(s * u, 0.f);
        actE[(size_t)m * I_ + n] = __builtin_bit_cast(__bf16, (unsigned short)pk);
      }
    }
  }
}

// ---------------------------------------------------------------------------
// Kernel 2: down GEMM.  act(bf16) [C x I] @ w_down(f32->bf16) [I x H].
// Grid (H/128, C/128, E); block tile 128x128; 8 waves 4x2, wave 32x64.
// ---------------------------------------------------------------------------
__global__ __launch_bounds__(256)
void gemm_down(const __bf16* __restrict__ act,
               const float* __restrict__ wd,
               float* __restrict__ out) {
  const int nblk = blockIdx.x;
  const int cblk = blockIdx.y;
  const int e    = blockIdx.z;

  constexpr int SA = 72;
  constexpr int SB = 72;
  __shared__ __align__(16) __bf16 AsB[2][128 * SA];
  __shared__ __align__(16) __bf16 BtB[2][128 * SB];

  const int tid  = threadIdx.x;
  const int lane = tid & 31;
  const int wave = tid >> 5;
  const int wm   = wave >> 1;
  const int wn   = wave & 1;

  const __bf16* actE = act + (size_t)e * C_ * I_ + (size_t)(cblk * 128) * I_;
  const float*  wdE  = wd  + (size_t)e * I_ * H_ + nblk * 128;

  v8f acc[2][4] = {};

  auto stage = [&](int k0, __bf16* As, __bf16* Bt) {
    // A: 128x64 bf16 tile, 16B copies (async direct-to-LDS when available)
    #pragma unroll
    for (int it = 0; it < 4; ++it) {
      int t   = tid + it * 256;
      int row = t >> 3;
      int c8  = (t & 7) << 3;
#if USE_ASYNC_LDS
      __builtin_amdgcn_global_load_async_to_lds_b128(
          to_gbl(actE + (size_t)row * I_ + k0 + c8),
          to_lds(&As[row * SA + c8]), 0, 0);
#else
      *(v8bf*)(&As[row * SA + c8]) =
          *(const v8bf*)(actE + (size_t)row * I_ + k0 + c8);
#endif
    }
    // B: 64k x 128n f32 -> bf16 transposed, k-pair packed b32 stores
    int kp = tid >> 5;          // 0..7
    int n4 = (tid & 31) << 2;   // 0..124
    #pragma unroll
    for (int kk = 0; kk < 32; kk += 8) {
      int k = (kp + kk) << 1;
      v4f w0 = *(const v4f*)(wdE + (size_t)(k0 + k) * H_ + n4);
      v4f w1 = *(const v4f*)(wdE + (size_t)(k0 + k + 1) * H_ + n4);
      #pragma unroll
      for (int i = 0; i < 4; ++i)
        *(unsigned*)(&Bt[(n4 + i) * SB + k]) = pk2bf(w0[i], w1[i]);
    }
  };

  stage(0, AsB[0], BtB[0]);

  for (int k0 = 0; k0 < I_; k0 += 64) {
    const int p = (k0 >> 6) & 1;
#if USE_ASYNC_LDS
    __builtin_amdgcn_s_wait_asynccnt(0);   // async LDS writes visible pre-barrier
#endif
    __syncthreads();
    if (k0 + 64 < I_) {
      stage(k0 + 64, AsB[p ^ 1], BtB[p ^ 1]);
      if (k0 + 128 < I_)
        __builtin_prefetch(wdE + (size_t)(k0 + 128 + (tid >> 3)) * H_ +
                               ((tid & 7) << 4), 0, 1);
    }
    const __bf16* As = AsB[p];
    const __bf16* Bt = BtB[p];
    #pragma unroll
    for (int kk = 0; kk < 2; ++kk) {
      v16bf af[2];
      #pragma unroll
      for (int mt = 0; mt < 2; ++mt) {
        int row = wm * 32 + mt * 16 + (lane & 15);
        int kb  = kk * 32 + ((lane >> 4) << 4);
        af[mt] = cat8(*(const v8bf*)(&As[row * SA + kb]),
                      *(const v8bf*)(&As[row * SA + kb + 8]));
      }
      #pragma unroll
      for (int nt = 0; nt < 4; ++nt) {
        int col = wn * 64 + nt * 16 + (lane & 15);
        int kb2 = kk * 32 + ((lane >> 4) << 4);
        v16bf bfrag = cat8(*(const v8bf*)(&Bt[col * SB + kb2]),
                           *(const v8bf*)(&Bt[col * SB + kb2 + 8]));
        #pragma unroll
        for (int mt = 0; mt < 2; ++mt) {
          acc[mt][nt] = __builtin_amdgcn_wmma_f32_16x16x32_bf16(
              false, af[mt], false, bfrag, (short)0, acc[mt][nt], false, false);
        }
      }
    }
  }

  const int mbase = cblk * 128 + wm * 32;
  const int nbase = nblk * 128 + wn * 64;
  float* outE = out + (size_t)e * C_ * H_;
  #pragma unroll
  for (int mt = 0; mt < 2; ++mt) {
    #pragma unroll
    for (int nt = 0; nt < 4; ++nt) {
      #pragma unroll
      for (int v = 0; v < 8; ++v) {
        int m = mbase + mt * 16 + v + ((lane >> 4) << 3);
        int n = nbase + nt * 16 + (lane & 15);
        outE[(size_t)m * H_ + n] = acc[mt][nt][v];
      }
    }
  }
}

// ---------------------------------------------------------------------------
extern "C" void kernel_launch(void* const* d_in, const int* in_sizes, int n_in,
                              void* d_out, int out_size, void* d_ws, size_t ws_size,
                              hipStream_t stream) {
  const float* hs  = (const float*)d_in[0];   // (E, C, H)
  const float* wgu = (const float*)d_in[1];   // (E, H, 2I)
  const float* wd  = (const float*)d_in[2];   // (E, I, H)
  float*  out = (float*)d_out;                // (E, C, H)
  __bf16* act = (__bf16*)d_ws;                // (E, C, I) bf16 scratch, 64 MB

  dim3 g1(I_ / 64, C_ / 128, E_);
  gemm_gate_up_silu<<<g1, 256, 0, stream>>>(hs, wgu, act);

  dim3 g2(H_ / 128, C_ / 128, E_);
  gemm_down<<<g2, 256, 0, stream>>>(act, wd, out);
}
